// Qwen2NSA_65128884076875
// MI455X (gfx1250) — matrix operand order repaired and checked by reading, not compiled, not run on hardware
//
#include <hip/hip_runtime.h>
#include <hip/hip_bf16.h>
#include <stdint.h>
#include <stddef.h>

// ---------------------------------------------------------------------------
// Qwen2 NSA forward for MI455X (gfx1250, wave32, WMMA bf16 + double-buffered
// Tensor Data Mover pipeline)
// ---------------------------------------------------------------------------

typedef __bf16 bf16_t;
typedef __attribute__((ext_vector_type(16))) __bf16 v16bf;
typedef __attribute__((ext_vector_type(8)))  float  v8f;
typedef __attribute__((ext_vector_type(4)))  uint32_t v4u;
typedef __attribute__((ext_vector_type(8)))  uint32_t v8u;

#define T_      2048
#define HID_    2048
#define HQ_     16
#define HKV_    2
#define D_      128
#define G_      8
#define KERNEL_ 32
#define SELECT_ 64
#define TOPN_   16
#define WIN_    512
#define NC_     64          // T / KERNEL
#define S_      32          // T / SELECT
#define SCALE_  0.08838834764831845f
#define NEGF    (-1.0e30f)

// ----------------------------- WMMA helpers --------------------------------

__device__ __forceinline__ v8f wmma_bf16(v16bf a, v16bf b, v8f c) {
  return __builtin_amdgcn_wmma_f32_16x16x32_bf16(false, a, false, b,
                                                 (short)0, c, false, false);
}

// A-matrix 16x32 (MxK) bf16 fragment. base points at (m=0, k=0), ld = row
// stride in elements. lane = m + 16*half ; ks = half*16 + {0..15}.
__device__ __forceinline__ v16bf load_a_frag(const bf16_t* base, int ld) {
  int lane = threadIdx.x & 31;
  int m  = lane & 15;
  int kb = (lane >> 4) << 4;
  const bf16_t* p = base + (size_t)m * ld + kb;
  v16bf f;
#pragma unroll
  for (int j = 0; j < 16; ++j) f[j] = p[j];
  return f;
}

// B-matrix 32x16 (KxN) where B^T is stored row-major (row n, contiguous K).
__device__ __forceinline__ v16bf load_b_frag_rowK(const bf16_t* base, int ld) {
  int lane = threadIdx.x & 31;
  int n  = lane & 15;
  int kb = (lane >> 4) << 4;
  const bf16_t* p = base + (size_t)n * ld + kb;
  v16bf f;
#pragma unroll
  for (int j = 0; j < 16; ++j) f[j] = p[j];
  return f;
}

// B-matrix 32x16 (KxN) stored K-major row-major (e.g. V: keys x D).
__device__ __forceinline__ v16bf load_b_frag_colK(const bf16_t* base, int ld) {
  int lane = threadIdx.x & 31;
  int n  = lane & 15;
  int kb = (lane >> 4) << 4;
  const bf16_t* p = base + (size_t)kb * ld + n;
  v16bf f;
#pragma unroll
  for (int j = 0; j < 16; ++j) f[j] = p[(size_t)j * ld];
  return f;
}

__device__ __forceinline__ v8f zero_v8f() {
  v8f z = {0.f, 0.f, 0.f, 0.f, 0.f, 0.f, 0.f, 0.f};
  return z;
}

// --------------------------- Tensor Data Mover -----------------------------
// Wave-level TDM DMA of a (rows x cols) bf16 tile with row stride ld_elems,
// into contiguous LDS at lds_byte_off. D# per CDNA5 ISA
// (2-D tensor, groups 2/3 NULL). Tracked with TENSORcnt.

__device__ __forceinline__ void tdm_load_tile_bf16(uint32_t lds_byte_off,
                                                   const bf16_t* gsrc,
                                                   uint32_t rows, uint32_t cols,
                                                   uint32_t ld_elems) {
  uint64_t ga = (uint64_t)(uintptr_t)gsrc;
  v4u g0;
  g0[0] = 1u;                                    // count=1, user descriptor
  g0[1] = lds_byte_off;                          // lds_addr
  g0[2] = (uint32_t)ga;                          // global_addr[31:0]
  g0[3] = (uint32_t)((ga >> 32) & 0x01ffffffu)   // global_addr[56:32]
          | (2u << 30);                          // type = 2 ("image")
  v8u g1;
  g1[0] = (1u << 16);                            // data_size=1 (2 bytes)
  g1[1] = (cols & 0xffffu) << 16;                // tensor_dim0[15:0]
  g1[2] = (cols >> 16) | ((rows & 0xffffu) << 16); // dim0[31:16], dim1[15:0]
  g1[3] = (rows >> 16) | (cols << 16);           // dim1[31:16], tile_dim0
  g1[4] = rows;                                  // tile_dim1 (tile_dim2 = 0)
  g1[5] = ld_elems;                              // tensor_dim0_stride[31:0]
  g1[6] = 0u;                                    // stride[47:32], dim1_stride lo
  g1[7] = 0u;
  asm volatile("tensor_load_to_lds %0, %1" :: "s"(g0), "s"(g1) : "memory");
}

// ----------------------------- fp32 -> bf16 --------------------------------

__global__ void cvt_bf16_kernel(const float* __restrict__ in,
                                bf16_t* __restrict__ out, int n) {
  int i = blockIdx.x * blockDim.x + threadIdx.x;
  if (i < n) out[i] = (bf16_t)in[i];
}

// --------------------- GEMM: C[M,N] = A[M,K] @ W[N,K]^T + bias -------------
// Tiles 128x128, K-step 32; 8 waves each owning 32x64. A/B tiles stream
// through ping-pong LDS buffers via the Tensor Data Mover: wave 0 DMAs A,
// wave 1 DMAs B, next tile's DMA is issued before computing the current one
// (wait TENSORcnt<=1), so the DMA engine runs ahead of the WMMA pipeline.

#define TM 128
#define TN 128
#define TK 32

__global__ __launch_bounds__(256) void gemm_bf16_wmma(
    const bf16_t* __restrict__ A,    // M x K  (bf16, row-major)
    const bf16_t* __restrict__ W,    // N x K  (bf16, row-major)
    const float*  __restrict__ bias, // N (nullable)
    float* __restrict__ C,           // M x N  (f32)
    int M, int N, int K) {
  __shared__ __align__(16) bf16_t sA[2][TM][TK];
  __shared__ __align__(16) bf16_t sB[2][TN][TK];

  int bm = blockIdx.x * TM;
  int bn = blockIdx.y * TN;
  int tid  = threadIdx.x;
  int lane = tid & 31;
  // Wave index as a *scalar* so the TDM issue branches are s_cbranch,
  // not v_cmpx/exec manipulation.
  int wave = __builtin_amdgcn_readfirstlane(tid) >> 5;
  int wm = wave & 3;   // 4 waves along M
  int wn = wave >> 2;  // 2 waves along N

  v8f acc[2][4];
#pragma unroll
  for (int i = 0; i < 2; ++i)
#pragma unroll
    for (int j = 0; j < 4; ++j) acc[i][j] = zero_v8f();

  uint32_t ldsA[2] = {(uint32_t)(uintptr_t)&sA[0][0][0],
                      (uint32_t)(uintptr_t)&sA[1][0][0]};
  uint32_t ldsB[2] = {(uint32_t)(uintptr_t)&sB[0][0][0],
                      (uint32_t)(uintptr_t)&sB[1][0][0]};
  const bf16_t* Abase = A + (size_t)bm * K;
  const bf16_t* Wbase = W + (size_t)bn * K;

  int nk = K / TK;
  // Prologue: DMA tile 0 into buffer 0.
  if (wave == 0) {
    tdm_load_tile_bf16(ldsA[0], Abase, TM, TK, (uint32_t)K);
  } else if (wave == 1) {
    tdm_load_tile_bf16(ldsB[0], Wbase, TN, TK, (uint32_t)K);
  }

  for (int i = 0; i < nk; ++i) {
    int cur = i & 1;
    bool hasNext = (i + 1) < nk;
    if (wave == 0) {
      if (hasNext) {
        tdm_load_tile_bf16(ldsA[cur ^ 1], Abase + (size_t)(i + 1) * TK,
                           TM, TK, (uint32_t)K);
        __builtin_amdgcn_s_wait_tensorcnt(1);   // tile i complete
      } else {
        __builtin_amdgcn_s_wait_tensorcnt(0);
      }
    } else if (wave == 1) {
      if (hasNext) {
        tdm_load_tile_bf16(ldsB[cur ^ 1], Wbase + (size_t)(i + 1) * TK,
                           TN, TK, (uint32_t)K);
        __builtin_amdgcn_s_wait_tensorcnt(1);
      } else {
        __builtin_amdgcn_s_wait_tensorcnt(0);
      }
    }
    __syncthreads();   // tile i visible to all waves

    v16bf a0 = load_a_frag(&sA[cur][wm * 32][0], TK);
    v16bf a1 = load_a_frag(&sA[cur][wm * 32 + 16][0], TK);
#pragma unroll
    for (int jn = 0; jn < 4; ++jn) {
      v16bf b = load_b_frag_rowK(&sB[cur][wn * 64 + jn * 16][0], TK);
      acc[0][jn] = wmma_bf16(a0, b, acc[0][jn]);
      acc[1][jn] = wmma_bf16(a1, b, acc[1][jn]);
    }
    __syncthreads();   // done reading buffer `cur`; it may be overwritten next
  }

  // Hoisted bias (one load per output column owned by this lane).
  float bcol[4] = {0.f, 0.f, 0.f, 0.f};
  if (bias) {
#pragma unroll
    for (int j = 0; j < 4; ++j)
      bcol[j] = bias[bn + wn * 64 + j * 16 + (lane & 15)];
  }

#pragma unroll
  for (int i = 0; i < 2; ++i)
#pragma unroll
    for (int j = 0; j < 4; ++j)
#pragma unroll
      for (int e = 0; e < 8; ++e) {
        int row = bm + wm * 32 + i * 16 + e + ((lane >> 4) << 3);
        int col = bn + wn * 64 + j * 16 + (lane & 15);
        C[(size_t)row * N + col] = acc[i][j][e] + bcol[j];
      }
}

// ------------------------------ RoPE (in place) ----------------------------

__global__ void rope_kernel(float* __restrict__ x, bf16_t* __restrict__ xbf,
                            const float* __restrict__ cs,
                            const float* __restrict__ sn, int heads) {
  int i = blockIdx.x * blockDim.x + threadIdx.x;
  int total = T_ * heads * (D_ / 2);
  if (i >= total) return;
  int d   = i % (D_ / 2);
  int h   = (i / (D_ / 2)) % heads;
  int t   = i / ((D_ / 2) * heads);
  size_t base = ((size_t)t * heads + h) * D_;
  float x0 = x[base + d];
  float x1 = x[base + d + 64];
  float c0 = cs[(size_t)t * D_ + d],      s0 = sn[(size_t)t * D_ + d];
  float c1 = cs[(size_t)t * D_ + d + 64], s1 = sn[(size_t)t * D_ + d + 64];
  float o0 = x0 * c0 - x1 * s0;
  float o1 = x1 * c1 + x0 * s1;
  x[base + d]      = o0;
  x[base + d + 64] = o1;
  xbf[base + d]      = (bf16_t)o0;
  xbf[base + d + 64] = (bf16_t)o1;
}

// ---------------------- KERNEL=32 compression of K/V -----------------------

__global__ void compress_kernel(const float* __restrict__ k,
                                const float* __restrict__ v,
                                const float* __restrict__ wck,
                                const float* __restrict__ wcv,
                                float* __restrict__ kcmp,
                                float* __restrict__ vcmp,
                                bf16_t* __restrict__ kcmp_bf,
                                bf16_t* __restrict__ vcmp_bf) {
  int i = blockIdx.x * blockDim.x + threadIdx.x;
  if (i >= NC_ * HKV_ * D_) return;
  int d = i % D_;
  int h = (i / D_) % HKV_;
  int n = i / (D_ * HKV_);
  float sk = 0.f, sv = 0.f;
#pragma unroll 4
  for (int j = 0; j < KERNEL_; ++j) {
    size_t idx = ((size_t)(n * KERNEL_ + j) * HKV_ + h) * D_ + d;
    sk += k[idx] * wck[j];
    sv += v[idx] * wcv[j];
  }
  kcmp[i] = sk; vcmp[i] = sv;
  kcmp_bf[i] = (bf16_t)sk; vcmp_bf[i] = (bf16_t)sv;
}

// ------------------- Compressed attention (+ p_cmp scores) -----------------

__global__ __launch_bounds__(32) void cmp_attn_kernel(
    const bf16_t* __restrict__ Qb,   // T x HQ x D
    const bf16_t* __restrict__ Kc,   // Nc x HKV x D
    const bf16_t* __restrict__ Vc,   // Nc x HKV x D
    float* __restrict__ Pcmp,        // HQ x T x Nc
    float* __restrict__ O) {         // T x HQ x D
  int h   = blockIdx.y;
  int t0  = blockIdx.x * 16;
  int hkv = h / G_;
  int lane = threadIdx.x;
  __shared__ __align__(16) bf16_t sP[16][NC_];

  v16bf qa[4];
#pragma unroll
  for (int c = 0; c < 4; ++c)
    qa[c] = load_a_frag(Qb + ((size_t)t0 * HQ_ + h) * D_ + c * 32, HQ_ * D_);

  v8f sc[4];
#pragma unroll
  for (int nt = 0; nt < 4; ++nt) sc[nt] = zero_v8f();

#pragma unroll
  for (int c = 0; c < 4; ++c)
#pragma unroll
    for (int nt = 0; nt < 4; ++nt) {
      v16bf kb = load_b_frag_rowK(
          Kc + ((size_t)(nt * 16) * HKV_ + hkv) * D_ + c * 32, HKV_ * D_);
      sc[nt] = wmma_bf16(qa[c], kb, sc[nt]);
    }

#pragma unroll
  for (int e = 0; e < 8; ++e) {
    int t = t0 + e + ((lane >> 4) << 3);
    float vals[4]; bool ok[4];
    float rowm = NEGF;
#pragma unroll
    for (int nt = 0; nt < 4; ++nt) {
      int n = nt * 16 + (lane & 15);
      float s = sc[nt][e] * SCALE_;
      ok[nt] = (t >= (n + 1) * KERNEL_ - 1);
      s = ok[nt] ? s : NEGF;
      vals[nt] = s;
      rowm = fmaxf(rowm, s);
    }
#pragma unroll
    for (int dmask = 8; dmask; dmask >>= 1)
      rowm = fmaxf(rowm, __shfl_xor(rowm, dmask, 16));
    float p[4]; float rs = 0.f;
#pragma unroll
    for (int nt = 0; nt < 4; ++nt) {
      p[nt] = ok[nt] ? __expf(vals[nt] - rowm) : 0.f;
      rs += p[nt];
    }
#pragma unroll
    for (int dmask = 8; dmask; dmask >>= 1)
      rs += __shfl_xor(rs, dmask, 16);
    float inv = (rs > 0.f) ? (1.f / rs) : 0.f;
    int r = e + ((lane >> 4) << 3);
#pragma unroll
    for (int nt = 0; nt < 4; ++nt) {
      float pv = p[nt] * inv;
      int n = nt * 16 + (lane & 15);
      Pcmp[((size_t)h * T_ + t) * NC_ + n] = pv;
      sP[r][n] = (bf16_t)pv;
    }
  }
  __syncthreads();

  v16bf pa0 = load_a_frag(&sP[0][0], NC_);
  v16bf pa1 = load_a_frag(&sP[0][32], NC_);
  v8f o[8];
#pragma unroll
  for (int j = 0; j < 8; ++j) o[j] = zero_v8f();
#pragma unroll
  for (int j = 0; j < 8; ++j) {
    v16bf v0 = load_b_frag_colK(Vc + ((size_t)0 * HKV_ + hkv) * D_ + j * 16,
                                HKV_ * D_);
    v16bf v1 = load_b_frag_colK(Vc + ((size_t)32 * HKV_ + hkv) * D_ + j * 16,
                                HKV_ * D_);
    o[j] = wmma_bf16(pa0, v0, o[j]);
    o[j] = wmma_bf16(pa1, v1, o[j]);
  }
#pragma unroll
  for (int j = 0; j < 8; ++j)
#pragma unroll
    for (int e = 0; e < 8; ++e) {
      int t = t0 + e + ((lane >> 4) << 3);
      O[((size_t)t * HQ_ + h) * D_ + j * 16 + (lane & 15)] = o[j][e];
    }
}

// ------------------------- Top-16 block selection --------------------------

__global__ void topk_kernel(const float* __restrict__ Pcmp,
                            unsigned* __restrict__ sel) {
  int idx = blockIdx.x * blockDim.x + threadIdx.x;
  if (idx >= HKV_ * T_) return;
  int hkv = idx / T_;
  int t   = idx % T_;
  float score[S_];
#pragma unroll
  for (int s = 0; s < S_; ++s) score[s] = 0.f;
  for (int g = 0; g < G_; ++g) {
    const float* pr = Pcmp + ((size_t)(hkv * G_ + g) * T_ + t) * NC_;
    for (int n = 0; n < NC_; ++n) score[n >> 1] += pr[n];
  }
  int cur = t / SELECT_;
#pragma unroll
  for (int s = 0; s < S_; ++s) {
    bool vis = (s * SELECT_) <= t;
    float bonus = (s == 0 || s == cur) ? 1.0e6f : 0.f;
    score[s] = vis ? (score[s] + bonus) : NEGF;
  }
  unsigned m = 0;
  for (int it = 0; it < TOPN_; ++it) {
    int best = 0; float bv = -3.0e38f;
    for (int s = 0; s < S_; ++s)
      if (!((m >> s) & 1u) && score[s] > bv) { bv = score[s]; best = s; }
    m |= (1u << best);
  }
  sel[idx] = m;
}

// ----------------- Masked flash attention (selected / window) --------------

__global__ __launch_bounds__(32) void attn_kernel(
    const bf16_t* __restrict__ Qb,   // T x HQ x D
    const bf16_t* __restrict__ Kb,   // T x HKV x D
    const bf16_t* __restrict__ Vb,   // T x HKV x D
    const unsigned* __restrict__ sel,// HKV x T block bitmasks
    float* __restrict__ O,           // T x HQ x D
    int mode) {                      // 0 = selected, 1 = window
  int h   = blockIdx.y;
  int t0  = blockIdx.x * 16;
  int hkv = h / G_;
  int lane = threadIdx.x;
  __shared__ __align__(16) bf16_t sP[16][32];

  v16bf qa[4];
#pragma unroll
  for (int c = 0; c < 4; ++c)
    qa[c] = load_a_frag(Qb + ((size_t)t0 * HQ_ + h) * D_ + c * 32, HQ_ * D_);

  v8f o[8];
  float mrow[8], lrow[8];
#pragma unroll
  for (int j = 0; j < 8; ++j) o[j] = zero_v8f();
#pragma unroll
  for (int e = 0; e < 8; ++e) { mrow[e] = NEGF; lrow[e] = 0.f; }

  unsigned selw[8];
#pragma unroll
  for (int e = 0; e < 8; ++e) {
    int t = t0 + e + ((lane >> 4) << 3);
    selw[e] = (mode == 0) ? sel[(size_t)hkv * T_ + t] : 0xffffffffu;
  }

  int ktEnd = (t0 + 15) >> 5;
  int ktBeg = 0;
  if (mode == 1) {
    int lo = t0 - WIN_ + 1;
    if (lo < 0) lo = 0;
    ktBeg = lo >> 5;
  }

  for (int kt = ktBeg; kt <= ktEnd; ++kt) {
    int s0 = kt * 32;
    if (kt + 1 <= ktEnd) {  // prefetch next K/V tile (global_prefetch_b8)
      __builtin_prefetch(Kb + ((size_t)(s0 + 32) * HKV_ + hkv) * D_, 0, 1);
      __builtin_prefetch(Vb + ((size_t)(s0 + 32) * HKV_ + hkv) * D_, 0, 1);
    }
    v8f sc0 = zero_v8f(), sc1 = zero_v8f();
#pragma unroll
    for (int c = 0; c < 4; ++c) {
      v16bf b0 = load_b_frag_rowK(
          Kb + ((size_t)s0 * HKV_ + hkv) * D_ + c * 32, HKV_ * D_);
      v16bf b1 = load_b_frag_rowK(
          Kb + ((size_t)(s0 + 16) * HKV_ + hkv) * D_ + c * 32, HKV_ * D_);
      sc0 = wmma_bf16(qa[c], b0, sc0);
      sc1 = wmma_bf16(qa[c], b1, sc1);
    }

#pragma unroll
    for (int e = 0; e < 8; ++e) {
      int t  = t0 + e + ((lane >> 4) << 3);
      int sA_ = s0 + (lane & 15);
      int sB_ = sA_ + 16;
      float v0 = sc0[e] * SCALE_;
      float v1 = sc1[e] * SCALE_;
      bool ok0, ok1;
      if (mode == 1) {
        ok0 = (sA_ <= t) && (t - sA_ < WIN_);
        ok1 = (sB_ <= t) && (t - sB_ < WIN_);
      } else {
        ok0 = (sA_ <= t) && ((selw[e] >> (sA_ / SELECT_)) & 1u);
        ok1 = (sB_ <= t) && ((selw[e] >> (sB_ / SELECT_)) & 1u);
      }
      v0 = ok0 ? v0 : NEGF;
      v1 = ok1 ? v1 : NEGF;
      float tmax = fmaxf(v0, v1);
#pragma unroll
      for (int dmask = 8; dmask; dmask >>= 1)
        tmax = fmaxf(tmax, __shfl_xor(tmax, dmask, 16));

      float mnew = fmaxf(mrow[e], tmax);
      float scale_old = __expf(mrow[e] - mnew);
      mrow[e] = mnew;
      float p0 = ok0 ? __expf(v0 - mnew) : 0.f;
      float p1 = ok1 ? __expf(v1 - mnew) : 0.f;
      float rs = p0 + p1;
#pragma unroll
      for (int dmask = 8; dmask; dmask >>= 1)
        rs += __shfl_xor(rs, dmask, 16);
      lrow[e] = lrow[e] * scale_old + rs;
#pragma unroll
      for (int j = 0; j < 8; ++j) o[j][e] *= scale_old;

      int r = e + ((lane >> 4) << 3);
      sP[r][lane & 15]        = (bf16_t)p0;
      sP[r][16 + (lane & 15)] = (bf16_t)p1;
    }
    __syncthreads();

    v16bf pa = load_a_frag(&sP[0][0], 32);
#pragma unroll
    for (int j = 0; j < 8; ++j) {
      v16bf vb = load_b_frag_colK(
          Vb + ((size_t)s0 * HKV_ + hkv) * D_ + j * 16, HKV_ * D_);
      o[j] = wmma_bf16(pa, vb, o[j]);
    }
    __syncthreads();
  }

#pragma unroll
  for (int e = 0; e < 8; ++e) {
    float inv = 1.f / fmaxf(lrow[e], 1e-20f);
    int t = t0 + e + ((lane >> 4) << 3);
#pragma unroll
    for (int j = 0; j < 8; ++j)
      O[((size_t)t * HQ_ + h) * D_ + j * 16 + (lane & 15)] = o[j][e] * inv;
  }
}

// ------------------------ Gate + combine the three paths -------------------

__global__ __launch_bounds__(128) void gate_combine_kernel(
    const float* __restrict__ q,     // T x HQ x D (roped)
    const float* __restrict__ Wg,    // HQ x D x 3
    const float* __restrict__ ocmp,
    const float* __restrict__ osel,
    const float* __restrict__ owin,
    float* __restrict__ ocomb,
    bf16_t* __restrict__ ocomb_bf) {
  int th = blockIdx.x;         // t*HQ + h
  int h  = th % HQ_;
  int d  = threadIdx.x;        // 0..127
  __shared__ float red[3][128];
  size_t i = (size_t)th * D_ + d;
  float qv = q[i];
#pragma unroll
  for (int e = 0; e < 3; ++e) red[e][d] = qv * Wg[(size_t)(h * D_ + d) * 3 + e];
  __syncthreads();
  for (int s = 64; s; s >>= 1) {
    if (d < s) {
#pragma unroll
      for (int e = 0; e < 3; ++e) red[e][d] += red[e][d + s];
    }
    __syncthreads();
  }
  float g0 = 1.f / (1.f + __expf(-red[0][0]));
  float g1 = 1.f / (1.f + __expf(-red[1][0]));
  float g2 = 1.f / (1.f + __expf(-red[2][0]));
  float v = g0 * ocmp[i] + g1 * osel[i] + g2 * owin[i];
  ocomb[i] = v;
  ocomb_bf[i] = (bf16_t)v;
}

// ---------------------------------------------------------------------------

static void* carve(char** p, size_t bytes) {
  void* r = *p;
  *p += (bytes + 255) & ~(size_t)255;
  return r;
}

extern "C" void kernel_launch(void* const* d_in, const int* in_sizes, int n_in,
                              void* d_out, int out_size, void* d_ws,
                              size_t ws_size, hipStream_t stream) {
  (void)in_sizes; (void)n_in; (void)out_size; (void)ws_size;
  const float* hs  = (const float*)d_in[0];
  const float* cs  = (const float*)d_in[1];
  const float* sn  = (const float*)d_in[2];
  const float* Wq  = (const float*)d_in[3];
  const float* bq  = (const float*)d_in[4];
  const float* Wk  = (const float*)d_in[5];
  const float* bk  = (const float*)d_in[6];
  const float* Wv  = (const float*)d_in[7];
  const float* bv  = (const float*)d_in[8];
  const float* Wo  = (const float*)d_in[9];
  const float* wck = (const float*)d_in[10];
  const float* wcv = (const float*)d_in[11];
  const float* Wg  = (const float*)d_in[12];
  float* out = (float*)d_out;

  char* wp = (char*)d_ws;
  float*  q_f    = (float*)carve(&wp, sizeof(float) * T_ * HQ_ * D_);
  float*  k_f    = (float*)carve(&wp, sizeof(float) * T_ * HKV_ * D_);
  float*  v_f    = (float*)carve(&wp, sizeof(float) * T_ * HKV_ * D_);
  float*  ocmp   = (float*)carve(&wp, sizeof(float) * T_ * HQ_ * D_);
  float*  osel   = (float*)carve(&wp, sizeof(float) * T_ * HQ_ * D_);
  float*  owin   = (float*)carve(&wp, sizeof(float) * T_ * HQ_ * D_);
  float*  ocomb  = (float*)carve(&wp, sizeof(float) * T_ * HQ_ * D_);
  float*  pcmp   = (float*)carve(&wp, sizeof(float) * HQ_ * T_ * NC_);
  float*  kcmp   = (float*)carve(&wp, sizeof(float) * NC_ * HKV_ * D_);
  float*  vcmp   = (float*)carve(&wp, sizeof(float) * NC_ * HKV_ * D_);
  unsigned* selm = (unsigned*)carve(&wp, sizeof(unsigned) * HKV_ * T_);
  bf16_t* h_bf   = (bf16_t*)carve(&wp, sizeof(bf16_t) * T_ * HID_);
  bf16_t* wq_bf  = (bf16_t*)carve(&wp, sizeof(bf16_t) * HQ_ * D_ * HID_);
  bf16_t* wk_bf  = (bf16_t*)carve(&wp, sizeof(bf16_t) * HKV_ * D_ * HID_);
  bf16_t* wv_bf  = (bf16_t*)carve(&wp, sizeof(bf16_t) * HKV_ * D_ * HID_);
  bf16_t* wo_bf  = (bf16_t*)carve(&wp, sizeof(bf16_t) * HID_ * HQ_ * D_);
  bf16_t* q_bf   = (bf16_t*)carve(&wp, sizeof(bf16_t) * T_ * HQ_ * D_);
  bf16_t* k_bf   = (bf16_t*)carve(&wp, sizeof(bf16_t) * T_ * HKV_ * D_);
  bf16_t* v_bf   = (bf16_t*)carve(&wp, sizeof(bf16_t) * T_ * HKV_ * D_);
  bf16_t* kcmp_b = (bf16_t*)carve(&wp, sizeof(bf16_t) * NC_ * HKV_ * D_);
  bf16_t* vcmp_b = (bf16_t*)carve(&wp, sizeof(bf16_t) * NC_ * HKV_ * D_);
  bf16_t* oc_bf  = (bf16_t*)carve(&wp, sizeof(bf16_t) * T_ * HQ_ * D_);

  auto cvt = [&](const float* src, bf16_t* dst, int n) {
    cvt_bf16_kernel<<<(n + 255) / 256, 256, 0, stream>>>(src, dst, n);
  };

  // 1) bf16 conversions of activations + weights
  cvt(hs, h_bf, T_ * HID_);
  cvt(Wq, wq_bf, HQ_ * D_ * HID_);
  cvt(Wk, wk_bf, HKV_ * D_ * HID_);
  cvt(Wv, wv_bf, HKV_ * D_ * HID_);
  cvt(Wo, wo_bf, HID_ * HQ_ * D_);

  // 2) QKV projections (WMMA, double-buffered TDM)
  gemm_bf16_wmma<<<dim3(T_ / TM, (HQ_ * D_) / TN), 256, 0, stream>>>(
      h_bf, wq_bf, bq, q_f, T_, HQ_ * D_, HID_);
  gemm_bf16_wmma<<<dim3(T_ / TM, (HKV_ * D_) / TN), 256, 0, stream>>>(
      h_bf, wk_bf, bk, k_f, T_, HKV_ * D_, HID_);
  gemm_bf16_wmma<<<dim3(T_ / TM, (HKV_ * D_) / TN), 256, 0, stream>>>(
      h_bf, wv_bf, bv, v_f, T_, HKV_ * D_, HID_);

  // 3) RoPE (in place) + bf16 copies; V just converted
  {
    int nq = T_ * HQ_ * (D_ / 2);
    rope_kernel<<<(nq + 255) / 256, 256, 0, stream>>>(q_f, q_bf, cs, sn, HQ_);
    int nk = T_ * HKV_ * (D_ / 2);
    rope_kernel<<<(nk + 255) / 256, 256, 0, stream>>>(k_f, k_bf, cs, sn, HKV_);
    cvt(v_f, v_bf, T_ * HKV_ * D_);
  }

  // 4) KERNEL=32 compression
  {
    int n = NC_ * HKV_ * D_;
    compress_kernel<<<(n + 255) / 256, 256, 0, stream>>>(
        k_f, v_f, wck, wcv, kcmp, vcmp, kcmp_b, vcmp_b);
  }

  // 5) Compressed attention + p_cmp
  cmp_attn_kernel<<<dim3(T_ / 16, HQ_), 32, 0, stream>>>(
      q_bf, kcmp_b, vcmp_b, pcmp, ocmp);

  // 6) Top-16 block selection bitmasks
  topk_kernel<<<(HKV_ * T_ + 127) / 128, 128, 0, stream>>>(pcmp, selm);

  // 7) Selected + window attention (flash-tiled, WMMA)
  attn_kernel<<<dim3(T_ / 16, HQ_), 32, 0, stream>>>(
      q_bf, k_bf, v_bf, selm, osel, 0);
  attn_kernel<<<dim3(T_ / 16, HQ_), 32, 0, stream>>>(
      q_bf, k_bf, v_bf, selm, owin, 1);

  // 8) Gate + combine
  gate_combine_kernel<<<T_ * HQ_, 128, 0, stream>>>(
      q_f, Wg, ocmp, osel, owin, ocomb, oc_bf);

  // 9) Output projection (WMMA) into d_out
  gemm_bf16_wmma<<<dim3(T_ / TM, HID_ / TN), 256, 0, stream>>>(
      oc_bf, wo_bf, nullptr, out, T_, HID_, HQ_ * D_);
}
